// SABlock_9826885174152
// MI455X (gfx1250) — matrix operand order repaired and checked by reading, not compiled, not run on hardware
//
#include <hip/hip_runtime.h>
#include <hip/hip_bf16.h>

// ---------------------------------------------------------------------------
// Problem constants (from the reference)
// ---------------------------------------------------------------------------
#define C_DIM      384
#define N_HEADS    6
#define HD         64            // head dim
#define HIDDEN_DIM 1536
#define B_SZ       2
#define DSP        14            // D = H = W = 14
#define NTOK       (DSP*DSP*DSP) // 2744 tokens per batch
#define M_TOK      (B_SZ*NTOK)   // 5488 total token rows
#define QKV_DIM    (3*C_DIM)     // 1152
#define NPAD       2752          // NTOK padded to multiple of 64
#define BH         (B_SZ*N_HEADS)

typedef __attribute__((ext_vector_type(16))) _Float16 v16h;
typedef __attribute__((ext_vector_type(8)))  _Float16 v8h;
typedef __attribute__((ext_vector_type(8)))  float    v8f;

// Concatenate two 8-half vectors into a WMMA 16-half operand.
__device__ __forceinline__ v16h cat8(v8h lo, v8h hh) {
    return __builtin_shufflevector(lo, hh, 0,1,2,3,4,5,6,7,8,9,10,11,12,13,14,15);
}
// A-operand fragment (16x32 f16): element e -> K = (e<8 ? e : e+8) + hi*8.
// Two contiguous 8-half runs at [hi*8] and [16+hi*8] of the lane's row.
__device__ __forceinline__ v16h afrag(const _Float16* row, int hi) {
    return cat8(*(const v8h*)(row + hi * 8), *(const v8h*)(row + 16 + hi * 8));
}
// B-operand fragment (32x16 f16): element e -> K = (e<8 ? e : e+8) + hi*8.
// Two contiguous 8-half runs at [hi*8] and [16+hi*8] of the lane's (column) row.
__device__ __forceinline__ v16h bfrag(const _Float16* row, int hi) {
    return cat8(*(const v8h*)(row + hi * 8), *(const v8h*)(row + 16 + hi * 8));
}

// ---------------------------------------------------------------------------
// 1) Depthwise 3x3x3 conv positional embedding + reshape to [M, C] (f32).
//    (conv_b is added and subtracted in the reference -> net no-op.)
// ---------------------------------------------------------------------------
__global__ void conv_pos_kernel(const float* __restrict__ x,
                                const float* __restrict__ cw,
                                float* __restrict__ t)
{
    int idx = blockIdx.x * blockDim.x + threadIdx.x;   // over B*C*NTOK
    if (idx >= B_SZ * C_DIM * NTOK) return;
    int n   = idx % NTOK;
    int tmp = idx / NTOK;
    int c   = tmp % C_DIM;
    int b   = tmp / C_DIM;
    int w_  = n % DSP;
    int h   = (n / DSP) % DSP;
    int d   = n / (DSP * DSP);
    const float* xb = x + ((size_t)(b * C_DIM + c)) * NTOK;
    float acc = xb[n];
    const float* wc = cw + c * 27;
    #pragma unroll
    for (int kd = -1; kd <= 1; ++kd)
        #pragma unroll
        for (int kh = -1; kh <= 1; ++kh)
            #pragma unroll
            for (int kw = -1; kw <= 1; ++kw) {
                int dd = d + kd, hh = h + kh, ww = w_ + kw;
                if (dd >= 0 && dd < DSP && hh >= 0 && hh < DSP && ww >= 0 && ww < DSP)
                    acc += wc[(kd + 1) * 9 + (kh + 1) * 3 + (kw + 1)] *
                           xb[(dd * DSP + hh) * DSP + ww];
            }
    t[((size_t)(b * NTOK + n)) * C_DIM + c] = acc;
}

// ---------------------------------------------------------------------------
// 2) LayerNorm over C=384 and cast to f16.
// ---------------------------------------------------------------------------
__global__ __launch_bounds__(128)
void ln_cast_kernel(const float* __restrict__ x,
                    const float* __restrict__ w,
                    const float* __restrict__ b,
                    _Float16* __restrict__ out)
{
    __shared__ float s1[128], s2[128];
    const int row = blockIdx.x;
    const int tid = threadIdx.x;
    const float* xr = x + (size_t)row * C_DIM;
    float a0 = xr[tid], a1 = xr[tid + 128], a2 = xr[tid + 256];
    s1[tid] = a0 + a1 + a2;
    s2[tid] = a0 * a0 + a1 * a1 + a2 * a2;
    __syncthreads();
    #pragma unroll
    for (int off = 64; off >= 1; off >>= 1) {
        if (tid < off) { s1[tid] += s1[tid + off]; s2[tid] += s2[tid + off]; }
        __syncthreads();
    }
    float mu  = s1[0] * (1.0f / C_DIM);
    float var = s2[0] * (1.0f / C_DIM) - mu * mu;
    float rs  = rsqrtf(var + 1e-5f);
    _Float16* orow = out + (size_t)row * C_DIM;
    orow[tid      ] = (_Float16)((a0 - mu) * rs * w[tid      ] + b[tid      ]);
    orow[tid + 128] = (_Float16)((a1 - mu) * rs * w[tid + 128] + b[tid + 128]);
    orow[tid + 256] = (_Float16)((a2 - mu) * rs * w[tid + 256] + b[tid + 256]);
}

// ---------------------------------------------------------------------------
// 3) f32 -> f16 cast (weights)
// ---------------------------------------------------------------------------
__global__ void cast_f16_kernel(const float* __restrict__ src,
                                _Float16* __restrict__ dst, int n)
{
    int i = blockIdx.x * blockDim.x + threadIdx.x;
    if (i < n) dst[i] = (_Float16)src[i];
}

// ---------------------------------------------------------------------------
// 4) Tiled WMMA GEMM: Y[M,N] = A[M,K](f16) * W[N,K]^T(f16) (+bias)(+epilogue)
//    Block: 128 threads = 4 waves; tile 128(M) x 64(N), K step 32.
//    Wave w owns rows w*32..w*32+31 (2 A-frags) x 64 cols -> 8 WMMAs/K-step.
//    All LDS tiles are fragment-contiguous:
//      As[row][k]  (stride 40 halfs = 80B: 16B aligned, conflict-free)
//      Bs[n][k]    (B fragment reads 8-k runs at hi*8 and 16+hi*8)
//    EPI: 0 = store f16, 1 = exact GELU -> f16, 2 = +resid -> f32 (in-place ok)
// ---------------------------------------------------------------------------
template<int EPI>
__global__ __launch_bounds__(128)
void gemm_wmma_kernel(const _Float16* __restrict__ A,
                      const _Float16* __restrict__ W,
                      const float* __restrict__ bias,
                      const float* resid,
                      void* outp,
                      int Mtot, int Nn, int Kk)
{
    __shared__ __align__(16) _Float16 As[128][40];
    __shared__ __align__(16) _Float16 Bs[64][40];
    const int tid  = threadIdx.x;
    const int lane = tid & 31;
    const int wave = tid >> 5;
    const int m0 = blockIdx.y * 128;
    const int n0 = blockIdx.x * 64;
    const int col = lane & 15;
    const int hi  = lane >> 4;
    const v8h zero8 = {0, 0, 0, 0, 0, 0, 0, 0};

    v8f acc[2][4] = {};

    for (int k0 = 0; k0 < Kk; k0 += 32) {
        // Stage A tile (128x32): thread t copies row t (32 halfs = 4x v8h).
        {
            int gm = m0 + tid;
            v8h* d8 = (v8h*)&As[tid][0];
            if (gm < Mtot) {
                const v8h* s8 = (const v8h*)(A + (size_t)gm * Kk + k0);
                d8[0] = s8[0]; d8[1] = s8[1]; d8[2] = s8[2]; d8[3] = s8[3];
            } else {
                d8[0] = zero8; d8[1] = zero8; d8[2] = zero8; d8[3] = zero8;
            }
        }
        // Stage W tile as Bs[n][k]: thread t -> n = t/2, 16 halfs of k.
        {
            int nl = tid >> 1, kb = (tid & 1) * 16;
            int gn = n0 + nl;
            v8h* d8 = (v8h*)&Bs[nl][kb];
            if (gn < Nn) {
                const v8h* s8 = (const v8h*)(W + (size_t)gn * Kk + k0 + kb);
                d8[0] = s8[0]; d8[1] = s8[1];
            } else {
                d8[0] = zero8; d8[1] = zero8;
            }
        }
        __syncthreads();

        v16h a[2];
        #pragma unroll
        for (int mt = 0; mt < 2; ++mt)
            a[mt] = afrag(&As[wave * 32 + mt * 16 + col][0], hi);
        #pragma unroll
        for (int nt = 0; nt < 4; ++nt) {
            v16h bf = bfrag(&Bs[nt * 16 + col][0], hi);
            #pragma unroll
            for (int mt = 0; mt < 2; ++mt)
                acc[mt][nt] = __builtin_amdgcn_wmma_f32_16x16x32_f16(
                    false, a[mt], false, bf, (short)0, acc[mt][nt], false, false);
        }
        __syncthreads();
    }

    // Epilogue (C layout: row = r + 8*hi, col = nt*16 + (lane&15))
    #pragma unroll
    for (int mt = 0; mt < 2; ++mt) {
        #pragma unroll
        for (int nt = 0; nt < 4; ++nt) {
            #pragma unroll
            for (int r = 0; r < 8; ++r) {
                int gm = m0 + wave * 32 + mt * 16 + r + 8 * hi;
                int gn = n0 + nt * 16 + col;
                if (gm < Mtot && gn < Nn) {
                    float v = acc[mt][nt][r];
                    if (bias) v += bias[gn];
                    size_t o = (size_t)gm * Nn + gn;
                    if constexpr (EPI == 0) {
                        ((_Float16*)outp)[o] = (_Float16)v;
                    } else if constexpr (EPI == 1) {
                        float g = 0.5f * v * (1.0f + erff(v * 0.70710678118654752f));
                        ((_Float16*)outp)[o] = (_Float16)g;
                    } else {
                        v += resid[o];
                        ((float*)outp)[o] = v;
                    }
                }
            }
        }
    }
}

// ---------------------------------------------------------------------------
// 5) Split fused QKV [M,1152] f16 into padded per-head buffers:
//      q [BH, NPAD, 64]  (pre-scaled by hd^-0.5 = 0.125)
//      k [BH, NPAD, 64]
//      vT[BH, 64, NPAD]  (transposed so attention V-fragments are contiguous)
//    Pad region (n >= NTOK) zero-filled -> attention staging needs no guards.
// ---------------------------------------------------------------------------
__global__ void qkv_pack_kernel(const _Float16* __restrict__ qkv,
                                _Float16* __restrict__ q,
                                _Float16* __restrict__ k,
                                _Float16* __restrict__ vT)
{
    int idx = blockIdx.x * blockDim.x + threadIdx.x;  // [BH, NPAD, 64] flat
    if (idx >= BH * NPAD * HD) return;
    int d   = idx & (HD - 1);
    int tmp = idx >> 6;
    int n   = tmp % NPAD;
    int bh  = tmp / NPAD;
    _Float16 qv = (_Float16)0.f, kv = (_Float16)0.f, vv = (_Float16)0.f;
    if (n < NTOK) {
        int b = bh / N_HEADS, h = bh % N_HEADS;
        size_t src = ((size_t)(b * NTOK + n)) * QKV_DIM + h * HD + d;
        qv = (_Float16)((float)qkv[src] * 0.125f);
        kv = qkv[src + C_DIM];
        vv = qkv[src + 2 * C_DIM];
    }
    q[idx] = qv;
    k[idx] = kv;
    vT[((size_t)bh * HD + d) * NPAD + n] = vv;
}

// ---------------------------------------------------------------------------
// 6) Flash-attention. Block = 128 threads = 4 waves; 64 query rows per block
//    (16 per wave), one batch-head per blockIdx.y. K-chunks of 32 keys staged
//    cooperatively once per chunk. Per wave per chunk:
//      S (16x32) = 2x2 WMMA, online softmax (width-16 shuffles),
//      P -> own LDS slab -> A-fragment, O (16x64) += 4 WMMA.
//    Output written as [M, C] f16 (proj GEMM A-operand layout).
// ---------------------------------------------------------------------------
__global__ __launch_bounds__(128)
void attn_kernel(const _Float16* __restrict__ q,
                 const _Float16* __restrict__ k,
                 const _Float16* __restrict__ vT,
                 _Float16* __restrict__ o)
{
    __shared__ __align__(16) _Float16 Qs[64][88];      // [qrow][d]
    __shared__ __align__(16) _Float16 Kn[32][88];      // [key][d]
    __shared__ __align__(16) _Float16 VT[64][40];      // [d][key]
    __shared__ __align__(16) _Float16 Ps[4][16][40];   // per-wave P slab

    const int tid  = threadIdx.x;
    const int lane = tid & 31;
    const int wave = tid >> 5;
    const int bh   = blockIdx.y;
    const int q0   = blockIdx.x * 64;
    const int col  = lane & 15;
    const int hi   = lane >> 4;
    const size_t baseQ = (size_t)bh * NPAD * HD;

    // Stage Q tile (64x64, already scaled): thread t -> row t/2, 32 halfs.
    {
        int r = tid >> 1, db = (tid & 1) * 32;
        const v8h* s8 = (const v8h*)(q + baseQ + (size_t)(q0 + r) * HD + db);
        v8h* d8 = (v8h*)&Qs[r][db];
        d8[0] = s8[0]; d8[1] = s8[1]; d8[2] = s8[2]; d8[3] = s8[3];
    }
    __syncthreads();

    v16h aq[2];
    #pragma unroll
    for (int s = 0; s < 2; ++s)
        aq[s] = afrag(&Qs[wave * 16 + col][s * 32], hi);

    v8f oacc[4] = {};
    float mrow[8], lrow[8];
    #pragma unroll
    for (int r = 0; r < 8; ++r) { mrow[r] = -3.0e38f; lrow[r] = 0.f; }

    const int nKC = NPAD / 32;   // 86 key chunks
    for (int j = 0; j < nKC; ++j) {
        __syncthreads();   // previous chunk's K/V fully consumed
        // Stage K chunk (32 keys x 64 d): thread t -> key t/4, 16 halfs of d.
        {
            int kk = tid >> 2, db = (tid & 3) * 16;
            const v8h* s8 = (const v8h*)(k + baseQ + (size_t)(j * 32 + kk) * HD + db);
            v8h* d8 = (v8h*)&Kn[kk][db];
            d8[0] = s8[0]; d8[1] = s8[1];
        }
        // Stage V^T chunk (64 d x 32 keys): thread t -> d t/2, 16 halfs of key.
        {
            int d = tid >> 1, kb = (tid & 1) * 16;
            const v8h* s8 = (const v8h*)(vT + ((size_t)bh * HD + d) * NPAD + j * 32 + kb);
            v8h* d8 = (v8h*)&VT[d][kb];
            d8[0] = s8[0]; d8[1] = s8[1];
        }
        __syncthreads();

        // S tiles: 16 queries x 32 keys, contraction over d=64 in 2 slices.
        v8f sacc[2] = {};
        #pragma unroll
        for (int t = 0; t < 2; ++t) {
            #pragma unroll
            for (int s = 0; s < 2; ++s) {
                v16h bk = bfrag(&Kn[t * 16 + col][s * 32], hi);
                sacc[t] = __builtin_amdgcn_wmma_f32_16x16x32_f16(
                    false, aq[s], false, bk, (short)0, sacc[t], false, false);
            }
        }
        // Mask out-of-range keys (padded region)
        #pragma unroll
        for (int t = 0; t < 2; ++t) {
            int gk = j * 32 + t * 16 + col;
            if (gk >= NTOK) {
                #pragma unroll
                for (int r = 0; r < 8; ++r) sacc[t][r] = -3.0e38f;
            }
        }
        // Online softmax per row (row M = r + 8*hi lives in one 16-lane half)
        #pragma unroll
        for (int r = 0; r < 8; ++r) {
            float mx = fmaxf(sacc[0][r], sacc[1][r]);
            #pragma unroll
            for (int off = 8; off >= 1; off >>= 1)
                mx = fmaxf(mx, __shfl_xor(mx, off, 16));
            float mnew  = fmaxf(mrow[r], mx);
            float alpha = __expf(mrow[r] - mnew);
            float p0 = __expf(sacc[0][r] - mnew);
            float p1 = __expf(sacc[1][r] - mnew);
            float ps = p0 + p1;
            #pragma unroll
            for (int off = 8; off >= 1; off >>= 1)
                ps += __shfl_xor(ps, off, 16);
            lrow[r] = lrow[r] * alpha + ps;
            mrow[r] = mnew;
            #pragma unroll
            for (int nt = 0; nt < 4; ++nt) oacc[nt][r] *= alpha;
            sacc[0][r] = p0;
            sacc[1][r] = p1;
        }
        // P (C layout) -> own LDS slab -> A-fragment (wave-local, no barrier)
        #pragma unroll
        for (int t = 0; t < 2; ++t)
            #pragma unroll
            for (int r = 0; r < 8; ++r)
                Ps[wave][r + 8 * hi][t * 16 + col] = (_Float16)sacc[t][r];
        v16h ap = afrag(&Ps[wave][col][0], hi);
        // O += P * V (contraction over 32 keys)
        #pragma unroll
        for (int nt = 0; nt < 4; ++nt) {
            v16h bv = bfrag(&VT[nt * 16 + col][0], hi);
            oacc[nt] = __builtin_amdgcn_wmma_f32_16x16x32_f16(
                false, ap, false, bv, (short)0, oacc[nt], false, false);
        }
    }

    // Normalize by l and scatter into [M, C] f16 layout
    const int b = bh / N_HEADS;
    const int h = bh % N_HEADS;
    #pragma unroll
    for (int r = 0; r < 8; ++r) {
        int gr = q0 + wave * 16 + r + 8 * hi;
        if (gr < NTOK) {
            float inv = 1.0f / lrow[r];
            #pragma unroll
            for (int nt = 0; nt < 4; ++nt) {
                int d = nt * 16 + col;
                o[((size_t)(b * NTOK + gr)) * C_DIM + h * HD + d] =
                    (_Float16)(oacc[nt][r] * inv);
            }
        }
    }
}

// ---------------------------------------------------------------------------
// 7) Final transpose [B,N,C] f32 -> [B,C,D,H,W] f32
// ---------------------------------------------------------------------------
__global__ void out_transpose_kernel(const float* __restrict__ t,
                                     float* __restrict__ out)
{
    int idx = blockIdx.x * blockDim.x + threadIdx.x;   // over B*C*NTOK
    if (idx >= B_SZ * C_DIM * NTOK) return;
    int n   = idx % NTOK;
    int tmp = idx / NTOK;
    int c   = tmp % C_DIM;
    int b   = tmp / C_DIM;
    out[idx] = t[((size_t)(b * NTOK + n)) * C_DIM + c];
}

// ---------------------------------------------------------------------------
// Host launcher
// ---------------------------------------------------------------------------
extern "C" void kernel_launch(void* const* d_in, const int* in_sizes, int n_in,
                              void* d_out, int out_size, void* d_ws, size_t ws_size,
                              hipStream_t stream)
{
    const float* x      = (const float*)d_in[0];
    const float* conv_w = (const float*)d_in[1];
    // conv_b (d_in[2]) nets out to zero contribution in the reference
    const float* ln1_w  = (const float*)d_in[3];
    const float* ln1_b  = (const float*)d_in[4];
    const float* qkv_w  = (const float*)d_in[5];
    const float* proj_w = (const float*)d_in[6];
    const float* proj_b = (const float*)d_in[7];
    const float* ln2_w  = (const float*)d_in[8];
    const float* ln2_b  = (const float*)d_in[9];
    const float* fc1_w  = (const float*)d_in[10];
    const float* fc1_b  = (const float*)d_in[11];
    const float* fc2_w  = (const float*)d_in[12];
    const float* fc2_b  = (const float*)d_in[13];
    float* outp = (float*)d_out;

    // ---- workspace carving (256B aligned) ----
    char* w8 = (char*)d_ws;
    size_t p = 0;
    auto take = [&](size_t bytes) -> char* {
        char* cur = w8 + p;
        p += (bytes + 255) & ~(size_t)255;
        return cur;
    };
    float*    t_f   = (float*)   take((size_t)M_TOK * C_DIM * 4);     // residual stream f32 (in-place updates)
    _Float16* tn    = (_Float16*)take((size_t)M_TOK * C_DIM * 2);     // LN1 out f16
    _Float16* wq16  = (_Float16*)take((size_t)QKV_DIM * C_DIM * 2);
    _Float16* wp16  = (_Float16*)take((size_t)C_DIM * C_DIM * 2);
    _Float16* w116  = (_Float16*)take((size_t)HIDDEN_DIM * C_DIM * 2);
    _Float16* w216  = (_Float16*)take((size_t)C_DIM * HIDDEN_DIM * 2);
    _Float16* qkv16 = (_Float16*)take((size_t)M_TOK * QKV_DIM * 2);
    _Float16* qh    = (_Float16*)take((size_t)BH * NPAD * HD * 2);
    _Float16* kh    = (_Float16*)take((size_t)BH * NPAD * HD * 2);
    _Float16* vTh   = (_Float16*)take((size_t)BH * HD * NPAD * 2);
    _Float16* o16   = (_Float16*)take((size_t)M_TOK * C_DIM * 2);     // attention out f16
    _Float16* tn2   = (_Float16*)take((size_t)M_TOK * C_DIM * 2);     // LN2 out f16
    _Float16* h16   = (_Float16*)take((size_t)M_TOK * HIDDEN_DIM * 2);// GELU out f16
    (void)ws_size; (void)in_sizes; (void)n_in; (void)out_size;

    const int elemTot = B_SZ * C_DIM * NTOK;     // 2,107,392

    // 1) conv pos-embed + tokenize
    conv_pos_kernel<<<(elemTot + 255) / 256, 256, 0, stream>>>(x, conv_w, t_f);

    // 2) cast weights to f16
    cast_f16_kernel<<<(QKV_DIM * C_DIM + 255) / 256, 256, 0, stream>>>(qkv_w, wq16, QKV_DIM * C_DIM);
    cast_f16_kernel<<<(C_DIM * C_DIM + 255) / 256, 256, 0, stream>>>(proj_w, wp16, C_DIM * C_DIM);
    cast_f16_kernel<<<(HIDDEN_DIM * C_DIM + 255) / 256, 256, 0, stream>>>(fc1_w, w116, HIDDEN_DIM * C_DIM);
    cast_f16_kernel<<<(C_DIM * HIDDEN_DIM + 255) / 256, 256, 0, stream>>>(fc2_w, w216, C_DIM * HIDDEN_DIM);

    // 3) LN1
    ln_cast_kernel<<<M_TOK, 128, 0, stream>>>(t_f, ln1_w, ln1_b, tn);

    const int mTiles = (M_TOK + 127) / 128;      // 43

    // 4) QKV GEMM [M,384]x[384,1152] -> f16
    gemm_wmma_kernel<0><<<dim3(QKV_DIM / 64, mTiles), 128, 0, stream>>>(
        tn, wq16, nullptr, nullptr, (void*)qkv16, M_TOK, QKV_DIM, C_DIM);

    // 5) split into padded q/k/vT per head (q pre-scaled)
    qkv_pack_kernel<<<(BH * NPAD * HD + 255) / 256, 256, 0, stream>>>(
        qkv16, qh, kh, vTh);

    // 6) flash attention: 43 query blocks x 12 batch-heads, 128 threads
    attn_kernel<<<dim3(NPAD / 64, BH), 128, 0, stream>>>(qh, kh, vTh, o16);

    // 7) proj GEMM + residual (in-place on t_f)
    gemm_wmma_kernel<2><<<dim3(C_DIM / 64, mTiles), 128, 0, stream>>>(
        o16, wp16, proj_b, t_f, (void*)t_f, M_TOK, C_DIM, C_DIM);

    // 8) LN2
    ln_cast_kernel<<<M_TOK, 128, 0, stream>>>(t_f, ln2_w, ln2_b, tn2);

    // 9) FC1 + exact GELU -> f16
    gemm_wmma_kernel<1><<<dim3(HIDDEN_DIM / 64, mTiles), 128, 0, stream>>>(
        tn2, w116, fc1_b, nullptr, (void*)h16, M_TOK, HIDDEN_DIM, C_DIM);

    // 10) FC2 + residual (in-place on t_f)
    gemm_wmma_kernel<2><<<dim3(C_DIM / 64, mTiles), 128, 0, stream>>>(
        h16, w216, fc2_b, t_f, (void*)t_f, M_TOK, C_DIM, HIDDEN_DIM);

    // 11) back to [B, C, D, H, W]
    out_transpose_kernel<<<(elemTot + 255) / 256, 256, 0, stream>>>(t_f, outp);
}
